// QwenStyleMoE_52424370815198
// MI455X (gfx1250) — compile-verified, matrix operand order checked by reading, not run on hardware
//
#include <hip/hip_runtime.h>
#include <hip/hip_bf16.h>
#include <math.h>
#include <stdint.h>

// ---------------------------------------------------------------------------
// Qwen-style MoE for MI455X (gfx1250, wave32, WMMA bf16 16x16x32, f32 acc).
// Memory-bound: 1.07 GB fp32 expert weights streamed ~once (~46us @23.3TB/s).
// f32 -> bf16 conversion in registers during LDS staging (no extra HBM bytes).
// bf16 operands staged with CDNA5 async global->LDS (ASYNCcnt) b128 copies.
// Branchless tile staging: A-tile rows are CLAMPED, not zero-padded, since
// D=A*B row m depends only on A row m and out-of-range rows are never stored.
// ---------------------------------------------------------------------------

typedef __attribute__((ext_vector_type(16))) __bf16 v16bf;
typedef __attribute__((ext_vector_type(8)))  float  v8f;

#define NTOK   2048   // B*T
#define DDIM   1024
#define HDIM   4096
#define HHALF  2048
#define NEXP   32
#define BM     128
#define BN     128
#define BK     32
#define LSTR   40     // LDS row stride in bf16 elems (80B: 16B-aligned, 64-bank conflict-free)

__device__ __forceinline__ unsigned short f2bf(float f) {
  unsigned u = __float_as_uint(f);
  u += 0x7FFFu + ((u >> 16) & 1u);     // round-to-nearest-even
  return (unsigned short)(u >> 16);
}
__device__ __forceinline__ unsigned pk2bf(float lo, float hi) {
  return (unsigned)f2bf(lo) | ((unsigned)f2bf(hi) << 16);
}
// v_rcp_f32-based sigmoid (~1 ulp): result is rounded to bf16 anyway, so a
// full IEEE divide expansion (div_scale/div_fmas) is wasted VALU work.
__device__ __forceinline__ float fast_sigmoid(float v) {
  return __builtin_amdgcn_rcpf(1.0f + __expf(-v));
}

union Frag { v16bf v; uint4 q[2]; };

// ---------------------------------------------------------------------------
// 128x128 block-tile GEMM: C[M,N] (+silu) = A[M,K] * B[K,N]
//  - A f32 (converted) or bf16 (async-to-LDS), optionally row-gathered
//  - B f32 row-major [K, ldb], converted to bf16, stored transposed in LDS
//  - 8 waves: wave (wm,wn) owns 32x64 -> 2x4 WMMA 16x16 f32 accumulators
// ---------------------------------------------------------------------------
template<bool A_BF16, bool GATHER, bool SILU>
__device__ __forceinline__ void gemm_tile(
    const void* __restrict__ Aany, int lda,
    const float* __restrict__ B, int ldb,
    void* __restrict__ Oany, int ldo,
    int Kdim, int M, int mbase, int nbase,
    const int* __restrict__ gat)
{
  __shared__ __align__(16) unsigned short As[BM * LSTR];
  __shared__ __align__(16) unsigned short Bs[BN * LSTR];
  __shared__ int toks[BM];

  const int tid  = threadIdx.x;
  const int lane = tid & 31;
  const int wid  = tid >> 5;
  const int wm   = wid & 3;   // 4 waves along M (32 rows each)
  const int wn   = wid >> 2;  // 2 waves along N (64 cols each)

  if (GATHER) {
    if (tid < BM) {
      int s = mbase + tid;
      if (s >= M) s = M - 1;          // clamp: padded rows are never stored
      toks[tid] = gat[s];
    }
    __syncthreads();
  }

  const v8f vzero = {0.f,0.f,0.f,0.f,0.f,0.f,0.f,0.f};
  v8f acc[2][4];
#pragma unroll
  for (int i = 0; i < 2; ++i)
#pragma unroll
    for (int j = 0; j < 4; ++j) acc[i][j] = vzero;

  const float*          Af = (const float*)Aany;
  const unsigned short* Ab = (const unsigned short*)Aany;

  // Per-thread staging coordinates (loop-invariant).
  const int ar   = tid >> 1;               // A: row 0..127 (2 threads/row)
  const int aseg = (tid & 1) << 4;         // A: k sub-segment 0 or 16
  const int agr  = mbase + ar;
  const int arc  = (agr < M) ? agr : (M - 1);   // clamped source row
  const int bp   = (tid >> 5) << 1;        // B: even k within 16-row half
  const int bn4  = (tid & 31) << 2;        // B: col group of 4

  for (int k0 = 0; k0 < Kdim; k0 += BK) {
    if (k0 + BK < Kdim) {  // hint next K-tile of weights
      __builtin_prefetch(&B[(size_t)(k0 + BK) * ldb + nbase + (tid & 127)], 0, 3);
    }

    // ---- stage A tile (BM x BK) into LDS as bf16 (branchless) ----
    if constexpr (A_BF16) {
      // async global->LDS copy of 32B row segment (two b128, shared IOFFSET
      // applies to both LDS and global sides).
      unsigned long long g =
          (unsigned long long)(Ab + (size_t)arc * lda + k0 + aseg);
      unsigned d = (unsigned)(uintptr_t)(&As[ar * LSTR + aseg]);
      asm volatile("global_load_async_to_lds_b128 %0, %1, off"
                   :: "v"(d), "v"(g) : "memory");
      asm volatile("global_load_async_to_lds_b128 %0, %1, off offset:16"
                   :: "v"(d), "v"(g) : "memory");
    } else {
      size_t row = GATHER ? (size_t)toks[ar] : (size_t)arc;
      const float4* src = (const float4*)(Af + row * lda + k0 + aseg);
      float4 v0 = src[0], v1 = src[1], v2 = src[2], v3 = src[3];
      uint4 u0, u1;
      u0.x = pk2bf(v0.x, v0.y); u0.y = pk2bf(v0.z, v0.w);
      u0.z = pk2bf(v1.x, v1.y); u0.w = pk2bf(v1.z, v1.w);
      u1.x = pk2bf(v2.x, v2.y); u1.y = pk2bf(v2.z, v2.w);
      u1.z = pk2bf(v3.x, v3.y); u1.w = pk2bf(v3.z, v3.w);
      *(uint4*)&As[ar * LSTR + aseg]     = u0;
      *(uint4*)&As[ar * LSTR + aseg + 8] = u1;
    }

    // ---- stage B tile (BK x BN) transposed -> Bs[n][k], f32->bf16 ----
    // Row-pair loads so adjacent k pack into one b32 LDS store.
#pragma unroll
    for (int pb = 0; pb < 2; ++pb) {
      int k = (pb << 4) + bp;            // even k in 0..30
      const float4* s0 = (const float4*)(B + (size_t)(k0 + k)     * ldb + nbase + bn4);
      const float4* s1 = (const float4*)(B + (size_t)(k0 + k + 1) * ldb + nbase + bn4);
      float4 a = *s0, b = *s1;
      *(unsigned*)&Bs[(bn4 + 0) * LSTR + k] = pk2bf(a.x, b.x);
      *(unsigned*)&Bs[(bn4 + 1) * LSTR + k] = pk2bf(a.y, b.y);
      *(unsigned*)&Bs[(bn4 + 2) * LSTR + k] = pk2bf(a.z, b.z);
      *(unsigned*)&Bs[(bn4 + 3) * LSTR + k] = pk2bf(a.w, b.w);
    }

    if constexpr (A_BF16) {
      asm volatile("s_wait_asynccnt 0x0" ::: "memory");  // ASYNCcnt != DScnt
    }
    __syncthreads();

    // ---- fragments per CDNA5 16-bit layout: lanes 0-15 K=0..7,16..23;
    //      lanes 16-31 K=8..15,24..31 (two ds_load_b128 per fragment) ----
    {
      int lm = lane & 15;
      int kh = (lane >> 4) << 3;   // 0 or 8
      Frag fa[2], fb[4];
#pragma unroll
      for (int mt = 0; mt < 2; ++mt) {
        const unsigned short* p = &As[(wm * 32 + mt * 16 + lm) * LSTR + kh];
        fa[mt].q[0] = *(const uint4*)(p);
        fa[mt].q[1] = *(const uint4*)(p + 16);
      }
#pragma unroll
      for (int nt = 0; nt < 4; ++nt) {
        const unsigned short* p = &Bs[(wn * 64 + nt * 16 + lm) * LSTR + kh];
        fb[nt].q[0] = *(const uint4*)(p);
        fb[nt].q[1] = *(const uint4*)(p + 16);
      }
#pragma unroll
      for (int mt = 0; mt < 2; ++mt)
#pragma unroll
        for (int nt = 0; nt < 4; ++nt)
          acc[mt][nt] = __builtin_amdgcn_wmma_f32_16x16x32_bf16(
              false, fa[mt].v, false, fb[nt].v, (short)0, acc[mt][nt],
              false, false);
    }
    __syncthreads();
  }

  // ---- epilogue: C layout lane n = L&15, m = r + 8*(L>=16) ----
  {
    int lm = lane & 15;
    int mh = (lane >> 4) << 3;
#pragma unroll
    for (int mt = 0; mt < 2; ++mt) {
#pragma unroll
      for (int nt = 0; nt < 4; ++nt) {
#pragma unroll
        for (int r = 0; r < 8; ++r) {
          int gm = mbase + wm * 32 + mt * 16 + mh + r;
          int gn = nbase + wn * 64 + nt * 16 + lm;
          if (gm < M) {
            float v = acc[mt][nt][r];
            if (SILU) {
              v = v * fast_sigmoid(v);               // silu = x*sigmoid(x)
              ((unsigned short*)Oany)[(size_t)gm * ldo + gn] = f2bf(v);
            } else {
              ((float*)Oany)[(size_t)gm * ldo + gn] = v;
            }
          }
        }
      }
    }
  }
}

// ---------------------------------------------------------------------------
// Kernel 1: routing. One wave per token, lane = expert (E==32==wave32).
// ---------------------------------------------------------------------------
__global__ void __launch_bounds__(256)
moe_gate(const float* __restrict__ x, const float* __restrict__ gw,
         const float* __restrict__ bias, int* __restrict__ tok_e,
         float* __restrict__ tok_w) {
  int wid  = threadIdx.x >> 5;
  int lane = threadIdx.x & 31;
  int t    = blockIdx.x * 8 + wid;
  if (t >= NTOK) return;

  const float4* xr = (const float4*)(x  + (size_t)t    * DDIM);
  const float4* gr = (const float4*)(gw + (size_t)lane * DDIM);
  float acc = 0.f;
  for (int i = 0; i < DDIM / 4; ++i) {
    float4 a = xr[i], b = gr[i];
    acc += a.x * b.x + a.y * b.y + a.z * b.z + a.w * b.w;
  }
  float s = fast_sigmoid(acc) + bias[lane];

  // top-1 (argmax, ties -> lowest index, matching jax.lax.top_k)
  float v = s; int idx = lane;
  for (int off = 16; off > 0; off >>= 1) {
    float ov = __shfl_xor(v, off, 32);
    int   oi = __shfl_xor(idx, off, 32);
    if (ov > v || (ov == v && oi < idx)) { v = ov; idx = oi; }
  }
  // top-2: mask winner, reduce again
  float s2 = (lane == idx) ? -3.4e38f : s;
  float v2 = s2; int idx2 = lane;
  for (int off = 16; off > 0; off >>= 1) {
    float ov = __shfl_xor(v2, off, 32);
    int   oi = __shfl_xor(idx2, off, 32);
    if (ov > v2 || (ov == v2 && oi < idx2)) { v2 = ov; idx2 = oi; }
  }
  if (lane == 0) {
    float inv = 1.0f / (v + v2 + 1e-20f);   // one real divide per token: fine
    tok_e[2 * t]     = idx;   tok_w[2 * t]     = v  * inv;
    tok_e[2 * t + 1] = idx2;  tok_w[2 * t + 1] = v2 * inv;
  }
}

// ---------------------------------------------------------------------------
// Kernel 2: ordered per-expert token lists via ballot + prefix popcount.
// ---------------------------------------------------------------------------
__global__ void moe_build_lists(const int* __restrict__ tok_e,
                                int* __restrict__ ecnt,
                                int* __restrict__ etok,
                                int* __restrict__ tok_pos) {
  int e    = blockIdx.x;
  int lane = threadIdx.x;
  int cnt  = 0;
  for (int base = 0; base < NTOK; base += 32) {
    int t  = base + lane;
    int e0 = tok_e[2 * t], e1 = tok_e[2 * t + 1];
    int kk = (e0 == e) ? 0 : ((e1 == e) ? 1 : -1);
    unsigned mask = (unsigned)__ballot(kk >= 0);   // wave32 -> low 32 bits
    int pre = __popc(mask & ((1u << lane) - 1u));
    if (kk >= 0) {
      etok[e * NTOK + cnt + pre] = t;
      tok_pos[2 * t + kk] = cnt + pre;
    }
    cnt += __popc(mask);
  }
  if (lane == 0) ecnt[e] = cnt;
}

// Kernel 3: exclusive scan of expert counts -> global slot offsets.
__global__ void moe_scan(const int* __restrict__ ecnt, int* __restrict__ eoff) {
  if (threadIdx.x == 0 && blockIdx.x == 0) {
    int o = 0;
    for (int e = 0; e < NEXP; ++e) { eoff[e] = o; o += ecnt[e]; }
    eoff[NEXP] = o;   // == NTOK*2 always
  }
}

// ---------------------------------------------------------------------------
// GEMM wrappers
// ---------------------------------------------------------------------------
__global__ void __launch_bounds__(256)
moe_gemm_shared1(const float* __restrict__ x, const float* __restrict__ w1,
                 unsigned short* __restrict__ h) {
  gemm_tile<false, false, true>(x, DDIM, w1, HHALF, h, HHALF,
                                DDIM, NTOK, blockIdx.y * BM, blockIdx.x * BN,
                                nullptr);
}

__global__ void __launch_bounds__(256)
moe_gemm_shared2(const unsigned short* __restrict__ h,
                 const float* __restrict__ w2, float* __restrict__ so) {
  gemm_tile<true, false, false>(h, HHALF, w2, DDIM, so, DDIM,
                                HHALF, NTOK, blockIdx.y * BM, blockIdx.x * BN,
                                nullptr);
}

__global__ void __launch_bounds__(256)
moe_expert_up(const float* __restrict__ x, const float* __restrict__ w1all,
              const int* __restrict__ ecnt, const int* __restrict__ eoff,
              const int* __restrict__ etok, unsigned short* __restrict__ hbuf) {
  int e     = blockIdx.z;
  int count = ecnt[e];
  int mbase = blockIdx.y * BM;
  if (mbase >= count) return;   // uniform early-exit, no divergent barrier
  const float* w1 = w1all + (size_t)e * DDIM * HDIM;
  unsigned short* hout = hbuf + (size_t)eoff[e] * HDIM;
  gemm_tile<false, true, true>(x, DDIM, w1, HDIM, hout, HDIM,
                               DDIM, count, mbase, blockIdx.x * BN,
                               etok + e * NTOK);
}

__global__ void __launch_bounds__(256)
moe_expert_down(const unsigned short* __restrict__ hbuf,
                const float* __restrict__ w2all,
                const int* __restrict__ ecnt, const int* __restrict__ eoff,
                float* __restrict__ ybuf) {
  int e     = blockIdx.z;
  int count = ecnt[e];
  int mbase = blockIdx.y * BM;
  if (mbase >= count) return;
  const float* w2 = w2all + (size_t)e * HDIM * DDIM;
  const unsigned short* hin = hbuf + (size_t)eoff[e] * HDIM;
  float* yout = ybuf + (size_t)eoff[e] * DDIM;
  gemm_tile<true, false, false>(hin, HDIM, w2, DDIM, yout, DDIM,
                                HDIM, count, mbase, blockIdx.x * BN, nullptr);
}

// ---------------------------------------------------------------------------
// Kernel 8: deterministic combine  out = shared + w0*y[slot0] + w1*y[slot1]
// ---------------------------------------------------------------------------
__global__ void __launch_bounds__(256)
moe_combine(const float* __restrict__ so, const float* __restrict__ ybuf,
            const int* __restrict__ tok_e, const float* __restrict__ tok_w,
            const int* __restrict__ tok_pos, const int* __restrict__ eoff,
            float* __restrict__ out) {
  int t  = blockIdx.x;
  float w0 = tok_w[2 * t], w1 = tok_w[2 * t + 1];
  int s0 = eoff[tok_e[2 * t]]     + tok_pos[2 * t];
  int s1 = eoff[tok_e[2 * t + 1]] + tok_pos[2 * t + 1];
  const float4* a = (const float4*)(so   + (size_t)t  * DDIM);
  const float4* b = (const float4*)(ybuf + (size_t)s0 * DDIM);
  const float4* c = (const float4*)(ybuf + (size_t)s1 * DDIM);
  float4* o = (float4*)(out + (size_t)t * DDIM);
  int i = threadIdx.x;                 // 256 threads == DDIM/4
  float4 va = a[i], vb = b[i], vc = c[i];
  float4 r;
  r.x = va.x + w0 * vb.x + w1 * vc.x;
  r.y = va.y + w0 * vb.y + w1 * vc.y;
  r.z = va.z + w0 * vb.z + w1 * vc.z;
  r.w = va.w + w0 * vb.w + w1 * vc.w;
  o[i] = r;
}

// ---------------------------------------------------------------------------
extern "C" void kernel_launch(void* const* d_in, const int* in_sizes, int n_in,
                              void* d_out, int out_size, void* d_ws,
                              size_t ws_size, hipStream_t stream) {
  const float* x    = (const float*)d_in[0];
  const float* gw   = (const float*)d_in[1];
  const float* bias = (const float*)d_in[2];
  const float* sw1  = (const float*)d_in[3];
  const float* sw2  = (const float*)d_in[4];
  const float* ew1  = (const float*)d_in[5];
  const float* ew2  = (const float*)d_in[6];
  float* out = (float*)d_out;

  char* ws = (char*)d_ws;
  size_t off = 0;
  auto take = [&](size_t bytes) -> char* {
    char* p = ws + off;
    off = (off + bytes + 255) & ~(size_t)255;
    return p;
  };
  int*   tok_e   = (int*)  take((size_t)NTOK * 2 * sizeof(int));
  float* tok_w   = (float*)take((size_t)NTOK * 2 * sizeof(float));
  int*   tok_pos = (int*)  take((size_t)NTOK * 2 * sizeof(int));
  int*   ecnt    = (int*)  take((size_t)NEXP * sizeof(int));
  int*   eoff    = (int*)  take((size_t)(NEXP + 1) * sizeof(int));
  int*   etok    = (int*)  take((size_t)NEXP * NTOK * sizeof(int));
  unsigned short* hsh  = (unsigned short*)take((size_t)NTOK * HHALF * 2); // 8 MB
  float*          so   = (float*)         take((size_t)NTOK * DDIM * 4); // 8 MB
  unsigned short* hbuf = (unsigned short*)take((size_t)NTOK * 2 * HDIM * 2); // 33.5 MB
  float*          ybuf = (float*)         take((size_t)NTOK * 2 * DDIM * 4); // 16.8 MB

  moe_gate<<<NTOK / 8, 256, 0, stream>>>(x, gw, bias, tok_e, tok_w);
  moe_build_lists<<<NEXP, 32, 0, stream>>>(tok_e, ecnt, etok, tok_pos);
  moe_scan<<<1, 32, 0, stream>>>(ecnt, eoff);
  moe_gemm_shared1<<<dim3(HHALF / BN, NTOK / BM), 256, 0, stream>>>(x, sw1, hsh);
  moe_gemm_shared2<<<dim3(DDIM / BN, NTOK / BM), 256, 0, stream>>>(hsh, sw2, so);
  moe_expert_up<<<dim3(HDIM / BN, NTOK / BM, NEXP), 256, 0, stream>>>(
      x, ew1, ecnt, eoff, etok, hbuf);
  moe_expert_down<<<dim3(DDIM / BN, NTOK / BM, NEXP), 256, 0, stream>>>(
      hbuf, ew2, ecnt, eoff, ybuf);
  moe_combine<<<NTOK, 256, 0, stream>>>(so, ybuf, tok_e, tok_w, tok_pos, eoff,
                                        out);
}
